// MultiHeadAttention_6493990552241
// MI455X (gfx1250) — matrix operand order repaired and checked by reading, not compiled, not run on hardware
//
#include <hip/hip_runtime.h>
#include <hip/hip_bf16.h>

// ---------------------------------------------------------------------------
// MHA forward for MI455X (gfx1250, wave32, WMMA bf16 path, async-LDS staging)
// B=4, S=2048, D=512, H=8, depth=64
// d_out = out[B,S,D] f32  ++  attn[B,H,S,S] f32
// ---------------------------------------------------------------------------

#define BB 4
#define SS 2048
#define DD 512
#define HH 8
#define DEPTH 64
#define PROW 2064            // padded LDS row stride for logits (2048+16)

typedef __bf16 v16bf __attribute__((ext_vector_type(16)));
typedef float  v8f   __attribute__((ext_vector_type(8)));

union ABReg { v16bf v; unsigned short us[16]; };

// A-matrix (16x32 bf16) K-offset for packed element j (ISA 7.12.2):
// VGPR v=j>>1; K = (v>=4 ? 16:0) + (v&3)*2 + (j&1)   [+ half*8 added by caller]
__device__ __forceinline__ int kOffA(int j) {
    int v = j >> 1;
    return ((v >> 2) << 4) | ((v & 3) << 1) | (j & 1);
}

__device__ __forceinline__ unsigned short bf16_bits(float f) {
    __bf16 h = (__bf16)f;                       // hardware v_cvt path
    return __builtin_bit_cast(unsigned short, h);
}

__device__ __forceinline__ v8f wmma_bf16(const v16bf& a, const v16bf& b, v8f c) {
    return __builtin_amdgcn_wmma_f32_16x16x32_bf16(false, a, false, b,
                                                   (short)0, c, false, false);
}

// CDNA5 async copy: 16 bytes global -> LDS, tracked by ASYNCcnt (ISA ch10/15.18)
__device__ __forceinline__ void async_copy_b128(unsigned lds_off, const float* gptr) {
    unsigned long long ga = (unsigned long long)gptr;
    asm volatile("global_load_async_to_lds_b128 %0, %1, off"
                 :: "v"(lds_off), "v"(ga) : "memory");
}
__device__ __forceinline__ void wait_asynccnt0() {
    asm volatile("s_wait_asynccnt 0" ::: "memory");
}
// LDS byte offset of a __shared__ object (flat shared addr keeps offset in low 32b)
__device__ __forceinline__ unsigned lds_off_of(const void* p) {
    return (unsigned)(unsigned long long)p;
}

// ---------------------------------------------------------------------------
// GEMM: Y = X[8192,512] @ W[512,512] + bias.
// Block tile 128(M) x 64(N), 8 waves each own 16 M-rows. K-step 32.
// X tile: async-copied to LDS as f32 (16KB). W tile: float4 load -> bf16 LDS
// (4KB), reused by all 8 waves. HEADSPLIT: bf16 out [B,H,S,64], else f32 [R,C].
// grid = (64, 8)
// ---------------------------------------------------------------------------
template<bool HEADSPLIT>
__global__ __launch_bounds__(256)
void gemm512_kernel(const float* __restrict__ x,
                    const float* __restrict__ w,
                    const float* __restrict__ bias,
                    void* __restrict__ outp) {
    __shared__ float          Xs[128 * 32];     // f32 staging (async target)
    __shared__ unsigned short Wsh[32 * 64];     // bf16 weight tile

    const int tid  = threadIdx.x;
    const int lane = tid & 31;
    const int wave = tid >> 5;
    const int hf   = lane >> 4;
    const int nl   = lane & 15;

    const int row0 = blockIdx.x * 128;
    const int n0   = blockIdx.y * 64;
    const int rowW = wave * 16;                 // this wave's M sub-tile in LDS

    const unsigned xsBase = lds_off_of(&Xs[0]);

    v8f acc[4];
    #pragma unroll
    for (int nn = 0; nn < 4; ++nn) acc[nn] = v8f{};

    for (int kb = 0; kb < DD; kb += 32) {
        __syncthreads();   // previous iteration's compute done before overwrite

        // ---- async-copy X tile 128x32 f32: 1024 float4, 4 per thread ----
        #pragma unroll
        for (int i = 0; i < 4; ++i) {
            const int idx = tid + i * 256;      // float4 index
            const int r   = idx >> 3;           // 0..127
            const int c4  = (idx & 7) << 2;     // 0,4,..,28
            async_copy_b128(xsBase + (unsigned)((r * 32 + c4) << 2),
                            &x[(size_t)(row0 + r) * DD + kb + c4]);
        }

        // ---- W tile 32x64: 512 float4, 2 per thread, convert to bf16 ----
        #pragma unroll
        for (int i = 0; i < 2; ++i) {
            const int idx = tid + i * 256;      // float4 index
            const int r   = idx >> 4;           // 0..31 (k)
            const int c   = (idx & 15) << 2;    // 0..60 (n)
            const float4 f = *(const float4*)&w[(size_t)(kb + r) * DD + n0 + c];
            Wsh[r * 64 + c + 0] = bf16_bits(f.x);
            Wsh[r * 64 + c + 1] = bf16_bits(f.y);
            Wsh[r * 64 + c + 2] = bf16_bits(f.z);
            Wsh[r * 64 + c + 3] = bf16_bits(f.w);
        }

        wait_asynccnt0();
        __syncthreads();

        // ---- WMMA: A 16x32 from Xs (cvt), B 32x16 x4 from Wsh ----
        ABReg a;
        #pragma unroll
        for (int j = 0; j < 16; ++j) {
            const int k = (hf << 4) + j;
            a.v[j] = (__bf16)Xs[(rowW + nl) * 32 + k];
        }
        #pragma unroll
        for (int nn = 0; nn < 4; ++nn) {
            ABReg bm;
            #pragma unroll
            for (int j = 0; j < 16; ++j) {
                const int k = (hf << 4) + j;    // B 32x16: K = half*16 + j
                bm.us[j] = Wsh[k * 64 + nn * 16 + nl];
            }
            acc[nn] = wmma_bf16(a.v, bm.v, acc[nn]);
        }
    }

    // C/D layout: VGPR r -> M = half*8 + r, N = lane&15
    #pragma unroll
    for (int nn = 0; nn < 4; ++nn) {
        #pragma unroll
        for (int r = 0; r < 8; ++r) {
            const int m = (hf << 3) + r;
            const int R = row0 + rowW + m;      // b*S + s
            const int C = n0 + nn * 16 + nl;    // h*64 + dd
            const float val = acc[nn][r] + bias[C];
            if (HEADSPLIT) {
                const int b = R >> 11, s = R & (SS - 1);
                const int h = C >> 6,  dd = C & 63;
                ((unsigned short*)outp)[(((size_t)(b * HH + h) * SS) + s) * DEPTH + dd] =
                    bf16_bits(val);
            } else {
                ((float*)outp)[(size_t)R * DD + C] = val;
            }
        }
    }
}

// ---------------------------------------------------------------------------
// Attention: one block per (b,h,16-query-row tile). 8 waves * 256 keys each.
// Logits+softmax in LDS, attn written once, then P@V via WMMA with LDS
// cross-wave reduction into f32 ctx [B,S,D].
// ---------------------------------------------------------------------------
__global__ __launch_bounds__(256)
void attn_kernel(const unsigned short* __restrict__ qh,
                 const unsigned short* __restrict__ kh,
                 const unsigned short* __restrict__ vh,
                 const int* __restrict__ mask,
                 float* __restrict__ attn_out,
                 float* __restrict__ ctx) {
    extern __shared__ float smem[];
    float* P    = smem;                 // [16][PROW] logits / probabilities
    float* RED  = smem + 16 * PROW;     // [16][16] softmax reductions (reused)
    float* PART = smem + 16 * PROW;     // [8][16*64] P@V partials (reused)

    const int tid  = threadIdx.x;
    const int lane = tid & 31;
    const int wave = tid >> 5;
    const int hf   = lane >> 4;
    const int nl   = lane & 15;

    const int qTile = blockIdx.x & 127;
    const int bh    = blockIdx.x >> 7;          // b*H + h
    const int b     = bh >> 3;
    const int h     = bh & 7;
    const size_t headBase = (size_t)bh * SS * DEPTH;
    const int q0 = qTile * 16;

    // ---- load this tile's Q as two A-matrices (16 x 32 bf16 each) ----
    ABReg qa[2];
    #pragma unroll
    for (int t = 0; t < 2; ++t) {
        #pragma unroll
        for (int j = 0; j < 16; ++j) {
            const int k = t * 32 + (hf << 4) + j;
            qa[t].us[j] = qh[headBase + (size_t)(q0 + nl) * DEPTH + k];
        }
    }

    // ---- phase 1: logits = scale * Q K^T + mask, into LDS ----
    const float scale = 0.125f;   // 1/sqrt(64)
    for (int kt = 0; kt < 16; ++kt) {
        const int key0 = wave * 256 + kt * 16;
        v8f acc = v8f{};
        #pragma unroll
        for (int t = 0; t < 2; ++t) {
            ABReg kb;                 // B 32(depth) x 16(keys)
            #pragma unroll
            for (int j = 0; j < 16; ++j) {
                const int dd = t * 32 + (hf << 4) + j;
                kb.us[j] = kh[headBase + (size_t)(key0 + nl) * DEPTH + dd];
            }
            acc = wmma_bf16(qa[t].v, kb.v, acc);
        }
        const int key = key0 + nl;
        const float mval = (float)mask[b * SS + key] * -1e9f;
        #pragma unroll
        for (int r = 0; r < 8; ++r) {
            const int m = (hf << 3) + r;
            P[m * PROW + key] = acc[r] * scale + mval;
        }
    }
    __syncthreads();

    // ---- phase 2: softmax over 2048 keys per row; 16 threads per row ----
    const int row = tid & 15;
    const int sub = tid >> 4;
    float* prow = P + row * PROW;

    float lm = -3.0e38f;
    for (int key = sub; key < SS; key += 16) lm = fmaxf(lm, prow[key]);
    RED[row * 16 + sub] = lm;
    __syncthreads();
    float rm = -3.0e38f;
    #pragma unroll
    for (int i = 0; i < 16; ++i) rm = fmaxf(rm, RED[row * 16 + i]);
    __syncthreads();

    float lsum = 0.0f;
    for (int key = sub; key < SS; key += 16) {
        float e = __expf(prow[key] - rm);
        prow[key] = e;
        lsum += e;
    }
    RED[row * 16 + sub] = lsum;
    __syncthreads();
    float rs = 0.0f;
    #pragma unroll
    for (int i = 0; i < 16; ++i) rs += RED[row * 16 + i];
    const float inv = 1.0f / rs;

    float* arow = attn_out + ((size_t)bh * SS + (q0 + row)) * SS;
    for (int key = sub; key < SS; key += 16) {
        const float p = prow[key] * inv;
        prow[key] = p;
        arow[key] = p;                 // materialize attn (reference output #2)
    }
    __syncthreads();

    // ---- phase 3: ctx partial = P[16 x my256keys] @ V[my256keys x 64] ----
    v8f c4[4];
    #pragma unroll
    for (int nn = 0; nn < 4; ++nn) c4[nn] = v8f{};

    for (int kt = 0; kt < 8; ++kt) {
        const int key0 = wave * 256 + kt * 32;
        ABReg a;                       // A 16(qrows) x 32(keys) from LDS probs
        #pragma unroll
        for (int j = 0; j < 16; ++j) {
            const int key = key0 + (hf << 4) + j;
            a.v[j] = (__bf16)P[nl * PROW + key];
        }
        #pragma unroll
        for (int nn = 0; nn < 4; ++nn) {
            ABReg bm;                  // B 32(keys) x 16(depth cols)
            #pragma unroll
            for (int j = 0; j < 16; ++j) {
                const int kk = key0 + (hf << 4) + j;
                bm.us[j] = vh[headBase + (size_t)kk * DEPTH + nn * 16 + nl];
            }
            c4[nn] = wmma_bf16(a.v, bm.v, c4[nn]);
        }
    }

    // cross-wave reduction of 16x64 partials through LDS
    #pragma unroll
    for (int nn = 0; nn < 4; ++nn) {
        #pragma unroll
        for (int r = 0; r < 8; ++r) {
            const int m = (hf << 3) + r;
            PART[wave * 1024 + m * 64 + nn * 16 + nl] = c4[nn][r];
        }
    }
    __syncthreads();
    for (int e = tid; e < 1024; e += 256) {
        float s = 0.0f;
        #pragma unroll
        for (int w = 0; w < 8; ++w) s += PART[w * 1024 + e];
        const int m = e >> 6, dd = e & 63;
        ctx[((size_t)(b * SS + q0 + m)) * DD + h * DEPTH + dd] = s;
    }
}

// ---------------------------------------------------------------------------
extern "C" void kernel_launch(void* const* d_in, const int* in_sizes, int n_in,
                              void* d_out, int out_size, void* d_ws, size_t ws_size,
                              hipStream_t stream) {
    (void)in_sizes; (void)n_in; (void)out_size; (void)ws_size;

    const float* q    = (const float*)d_in[0];
    const float* k    = (const float*)d_in[1];
    const float* v    = (const float*)d_in[2];
    const int*   mask = (const int*)  d_in[3];
    const float* wq   = (const float*)d_in[4];
    const float* bq   = (const float*)d_in[5];
    const float* wk   = (const float*)d_in[6];
    const float* bk   = (const float*)d_in[7];
    const float* wv   = (const float*)d_in[8];
    const float* bv   = (const float*)d_in[9];
    const float* wo   = (const float*)d_in[10];
    const float* bo   = (const float*)d_in[11];

    float* out  = (float*)d_out;
    float* attn = out + (size_t)BB * SS * DD;       // tuple output #2

    const size_t nElem = (size_t)BB * SS * DD;      // 4,194,304
    unsigned short* qh = (unsigned short*)d_ws;     // bf16 head-split
    unsigned short* kh = qh + nElem;
    unsigned short* vh = kh + nElem;
    float*          ctx = (float*)(vh + nElem);     // f32 [B,S,D]

    dim3 blk(256);
    dim3 ggrid(64, 8);

    gemm512_kernel<true ><<<ggrid, blk, 0, stream>>>(q, wq, bq, (void*)qh);
    gemm512_kernel<true ><<<ggrid, blk, 0, stream>>>(k, wk, bk, (void*)kh);
    gemm512_kernel<true ><<<ggrid, blk, 0, stream>>>(v, wv, bv, (void*)vh);

    const size_t shmem = (size_t)(16 * PROW + 8 * 1024) * sizeof(float); // ~165 KB
    (void)hipFuncSetAttribute((const void*)attn_kernel,
                              hipFuncAttributeMaxDynamicSharedMemorySize,
                              (int)shmem);
    attn_kernel<<<dim3(BB * HH * (SS / 16)), blk, shmem, stream>>>(
        qh, kh, vh, mask, attn, ctx);

    gemm512_kernel<false><<<ggrid, blk, 0, stream>>>(ctx, wo, bo, (void*)out);
}
